// Model_22316650070497
// MI455X (gfx1250) — matrix operand order repaired and checked by reading, not compiled, not run on hardware
//
#include <hip/hip_runtime.h>
#include <math.h>

#define NHEADS 16
#define DHEAD  64
#define BB     4
#define NN     1024
#define DM     1024
#define D2     2048   // 2*DHEAD*NHEADS

typedef __attribute__((ext_vector_type(16))) __bf16 v16bf;
typedef __attribute__((ext_vector_type(8)))  __bf16 v8bf;
typedef __attribute__((ext_vector_type(8)))  float  v8f;
typedef int v4i __attribute__((vector_size(16)));

#if __has_builtin(__builtin_amdgcn_global_load_async_to_lds_b128) && \
    __has_builtin(__builtin_amdgcn_s_wait_asynccnt)
#define ASYNC_COPY 1
#else
#define ASYNC_COPY 0
#endif

typedef __attribute__((address_space(1))) void as1_void;
typedef __attribute__((address_space(3))) void as3_void;
typedef __attribute__((address_space(1))) v4i  as1_v4i;
typedef __attribute__((address_space(3))) v4i  as3_v4i;

static __device__ inline void cp16(__bf16* l, const __bf16* g) {
#if ASYNC_COPY
  __builtin_amdgcn_global_load_async_to_lds_b128(
      (as1_v4i*)(as1_void*)(void*)(g),
      (as3_v4i*)(as3_void*)(void*)(l), 0, 0);
#else
  *(uint4*)(void*)(l) = *(const uint4*)(const void*)(g);
#endif
}

static __device__ inline v8f wmma_bf(v16bf a, v16bf b, v8f c) {
  return __builtin_amdgcn_wmma_f32_16x16x32_bf16(false, a, false, b, (short)0, c,
                                                 false, false);
}

// A-matrix 16x32 bf16 fragment: lane = (lo,hi); row = lo, this lane holds
// K = hi*16 + [0..15] (contiguous, matching the B fragment).
static __device__ inline v16bf load_a_frag(const __bf16* row, int hi) {
  v8bf p0 = *(const v8bf*)(row + hi * 16);
  v8bf p1 = *(const v8bf*)(row + hi * 16 + 8);
  v16bf a;
#pragma unroll
  for (int i = 0; i < 8; ++i) { a[i] = p0[i]; a[8 + i] = p1[i]; }
  return a;
}

// B-matrix 32x16 bf16 fragment: lane = (lo,hi); col = lo, lane holds
// K = hi*16 + [0..15], contiguous in memory (caller adds hi*16).
static __device__ inline v16bf load_b_frag(const __bf16* p) {
  v8bf p0 = *(const v8bf*)(p);
  v8bf p1 = *(const v8bf*)(p + 8);
  v16bf b;
#pragma unroll
  for (int i = 0; i < 8; ++i) { b[i] = p0[i]; b[8 + i] = p1[i]; }
  return b;
}

static __device__ inline float redmax16(float v) {
  v = fmaxf(v, __shfl_xor(v, 1));
  v = fmaxf(v, __shfl_xor(v, 2));
  v = fmaxf(v, __shfl_xor(v, 4));
  v = fmaxf(v, __shfl_xor(v, 8));
  return v;
}
static __device__ inline float redsum16(float v) {
  v += __shfl_xor(v, 1);
  v += __shfl_xor(v, 2);
  v += __shfl_xor(v, 4);
  v += __shfl_xor(v, 8);
  return v;
}

// ---------------------------------------------------------------- converts
__global__ void f32_to_bf16_kernel(const float* __restrict__ in,
                                   __bf16* __restrict__ out, int n) {
  int i = blockIdx.x * 256 + threadIdx.x;
  if (i < n) out[i] = (__bf16)in[i];
}

// lam[h] = exp(dot(lq1,lk1)) - exp(dot(lq2,lk2)) + 0.8
__global__ void lam_kernel(const float* __restrict__ lq1, const float* __restrict__ lk1,
                           const float* __restrict__ lq2, const float* __restrict__ lk2,
                           float* __restrict__ lam) {
  int h = threadIdx.x;
  if (h >= NHEADS) return;
  float d1 = 0.f, d2 = 0.f;
  for (int d = 0; d < DHEAD; ++d) {
    d1 += lq1[h * DHEAD + d] * lk1[h * DHEAD + d];
    d2 += lq2[h * DHEAD + d] * lk2[h * DHEAD + d];
  }
  lam[h] = expf(d1) - expf(d2) + 0.8f;
}

// Vt[(b*16+h)*128 + v][key] = Vp[(b*1024+key)*2048 + h*128 + v]
__global__ void transpose_v_kernel(const __bf16* __restrict__ Vp,
                                   __bf16* __restrict__ Vt) {
  int idx = blockIdx.x * 256 + threadIdx.x;
  if (idx >= BB * NHEADS * 2 * DHEAD * NN) return;
  int key = idx & (NN - 1);
  int v   = (idx >> 10) & 127;
  int h   = (idx >> 17) & 15;
  int b   = idx >> 21;
  Vt[idx] = Vp[((size_t)(b * NN + key)) * D2 + h * 2 * DHEAD + v];
}

// ---------------------------------------------------------------- NT GEMM
// C[M,N] = A[M,K] (row-major bf16) @ B[N,K]^T (row-major bf16)
// block = 256 threads = 8 waves (2 x 4), wave tile 64x64 -> block tile 128x256.
// A/B tiles double-buffered in LDS, filled with async global->LDS b128 copies.
template <bool OUT_BF16>
__global__ __launch_bounds__(256) void gemm_nt_kernel(
    const __bf16* __restrict__ A, const __bf16* __restrict__ B,
    float* __restrict__ Cf, __bf16* __restrict__ Cb, int M, int N, int K) {
  __shared__ __bf16 sA[2][128 * 32];  // 2 x 8 KB
  __shared__ __bf16 sB[2][256 * 32];  // 2 x 16 KB

  const int lane = threadIdx.x & 31;
  const int lo = lane & 15, hi = lane >> 4;
  const int w = threadIdx.x >> 5;
  const int wm = w >> 2, wn = w & 3;
  const int m0b = blockIdx.y * 128;
  const int n0b = blockIdx.x * 256;

  // each thread copies 2 x 16B of A and 4 x 16B of B per K-step
  auto issue = [&](int buf, int kk) {
#pragma unroll
    for (int i = 0; i < 2; ++i) {
      int c = threadIdx.x + i * 256;           // 0..511
      int row = c >> 2, q = c & 3;
      cp16(&sA[buf][row * 32 + q * 8], A + (size_t)(m0b + row) * K + kk + q * 8);
    }
#pragma unroll
    for (int i = 0; i < 4; ++i) {
      int c = threadIdx.x + i * 256;           // 0..1023
      int row = c >> 2, q = c & 3;
      cp16(&sB[buf][row * 32 + q * 8], B + (size_t)(n0b + row) * K + kk + q * 8);
    }
  };

  v8f acc[4][4] = {};

  issue(0, 0);
  for (int k = 0; k < K; k += 32) {
    const int buf = (k >> 5) & 1;
    if (k + 32 < K) {
      issue(buf ^ 1, k + 32);
#if ASYNC_COPY
      __builtin_amdgcn_s_wait_asynccnt(6);   // previous batch landed (in-order)
#endif
    } else {
#if ASYNC_COPY
      __builtin_amdgcn_s_wait_asynccnt(0);
#endif
    }
    __syncthreads();

    v16bf af[4], bfr[4];
#pragma unroll
    for (int i = 0; i < 4; ++i)
      af[i] = load_a_frag(&sA[buf][(wm * 64 + i * 16 + lo) * 32], hi);
#pragma unroll
    for (int j = 0; j < 4; ++j)
      bfr[j] = load_b_frag(&sB[buf][(wn * 64 + j * 16 + lo) * 32 + hi * 16]);
#pragma unroll
    for (int i = 0; i < 4; ++i)
#pragma unroll
      for (int j = 0; j < 4; ++j)
        acc[i][j] = wmma_bf(af[i], bfr[j], acc[i][j]);

    __syncthreads();  // everyone done reading buf before it is refilled
  }

#pragma unroll
  for (int i = 0; i < 4; ++i)
#pragma unroll
    for (int j = 0; j < 4; ++j)
#pragma unroll
      for (int r = 0; r < 8; ++r) {
        size_t row = (size_t)(m0b + wm * 64 + i * 16 + r + hi * 8);
        size_t col = (size_t)(n0b + wn * 64 + j * 16 + lo);
        if (OUT_BF16) Cb[row * N + col] = (__bf16)acc[i][j][r];
        else          Cf[row * N + col] = acc[i][j][r];
      }
}

// ---------------------------------------------------------------- attention
// One wave per (b,h, 16-query block). Flash-style dual online softmax.
__global__ __launch_bounds__(32) void diff_attn_kernel(
    const __bf16* __restrict__ Qp, const __bf16* __restrict__ Kp,
    const __bf16* __restrict__ Vt, const float* __restrict__ lam,
    const float* __restrict__ rms_scale, __bf16* __restrict__ Ocat) {
  __shared__ __bf16 pst[2][16 * 32];  // probabilities staging (D-layout -> A-layout)

  const int lane = threadIdx.x;
  const int lo = lane & 15, hi = lane >> 4;
  const int qb = blockIdx.x;            // query block (16 queries)
  const int bh = blockIdx.y;            // b*16+h
  const int b = bh >> 4, h = bh & 15;
  const int q0 = qb * 16;
  const float scaling = 0.125f;         // 1/sqrt(64)

  const __bf16* Qrow  = Qp + (size_t)(b * NN + q0 + lo) * D2 + h * 2 * DHEAD;
  const __bf16* Kbase = Kp + (size_t)(b * NN) * D2 + h * 2 * DHEAD;
  const __bf16* Vtb   = Vt + (size_t)bh * 2 * DHEAD * NN;

  // Q fragments (A-matrix): dh split into two K=32 chunks; Q1 = dims 0..63, Q2 = 64..127
  v16bf q1f0 = load_a_frag(Qrow + 0,  hi);
  v16bf q1f1 = load_a_frag(Qrow + 32, hi);
  v16bf q2f0 = load_a_frag(Qrow + 64, hi);
  v16bf q2f1 = load_a_frag(Qrow + 96, hi);

  v8f o1[8] = {};  // 16q x 128v accumulators, softmax #1
  v8f o2[8] = {};  // softmax #2
  float mx1[8], mx2[8], l1[8], l2[8];
#pragma unroll
  for (int r = 0; r < 8; ++r) { mx1[r] = mx2[r] = -1e30f; l1[r] = l2[r] = 0.f; }

  const int kend = q0 + 16;  // causal: keys < kend
  for (int kc = 0; kc < kend; kc += 32) {
    // ---- scores: two 16x16 tiles per set, contraction dh=64 (2 wmma each)
    v8f s1[2] = {};
    v8f s2[2] = {};
#pragma unroll
    for (int t = 0; t < 2; ++t) {
      const __bf16* Krow = Kbase + (size_t)(kc + t * 16 + lo) * D2;
      s1[t] = wmma_bf(q1f0, load_b_frag(Krow + 0  + hi * 16), s1[t]);
      s1[t] = wmma_bf(q1f1, load_b_frag(Krow + 32 + hi * 16), s1[t]);
      s2[t] = wmma_bf(q2f0, load_b_frag(Krow + 64 + hi * 16), s2[t]);
      s2[t] = wmma_bf(q2f1, load_b_frag(Krow + 96 + hi * 16), s2[t]);
    }

    // ---- mask + dual online softmax; rows live at reg r, lanes of half-wave
#pragma unroll
    for (int r = 0; r < 8; ++r) {
      int q = q0 + r + hi * 8;
      int k0i = kc + lo, k1i = kc + 16 + lo;

      float a0 = s1[0][r] * scaling; if (k0i > q) a0 = -1e30f;
      float a1 = s1[1][r] * scaling; if (k1i > q) a1 = -1e30f;
      float rm = redmax16(fmaxf(a0, a1));
      float mn = fmaxf(mx1[r], rm);
      float sc = __expf(mx1[r] - mn);
      float p0 = __expf(a0 - mn), p1v = __expf(a1 - mn);
      l1[r] = l1[r] * sc + redsum16(p0 + p1v);
      mx1[r] = mn;
#pragma unroll
      for (int t = 0; t < 8; ++t) o1[t][r] *= sc;
      pst[0][(r + hi * 8) * 32 + lo]      = (__bf16)p0;
      pst[0][(r + hi * 8) * 32 + 16 + lo] = (__bf16)p1v;

      float b0 = s2[0][r] * scaling; if (k0i > q) b0 = -1e30f;
      float b1 = s2[1][r] * scaling; if (k1i > q) b1 = -1e30f;
      float rm2 = redmax16(fmaxf(b0, b1));
      float mn2 = fmaxf(mx2[r], rm2);
      float sc2 = __expf(mx2[r] - mn2);
      float r0 = __expf(b0 - mn2), r1 = __expf(b1 - mn2);
      l2[r] = l2[r] * sc2 + redsum16(r0 + r1);
      mx2[r] = mn2;
#pragma unroll
      for (int t = 0; t < 8; ++t) o2[t][r] *= sc2;
      pst[1][(r + hi * 8) * 32 + lo]      = (__bf16)r0;
      pst[1][(r + hi * 8) * 32 + 16 + lo] = (__bf16)r1;
    }

    // ---- P (16x32, via LDS relayout) @ V-chunk (32x128) -> 8 tiles x 2 sets
    v16bf p1 = load_a_frag(&pst[0][lo * 32], hi);
    v16bf p2 = load_a_frag(&pst[1][lo * 32], hi);
#pragma unroll
    for (int t = 0; t < 8; ++t) {
      v16bf vb = load_b_frag(Vtb + (size_t)(t * 16 + lo) * NN + kc + hi * 16);
      o1[t] = wmma_bf(p1, vb, o1[t]);
      o2[t] = wmma_bf(p2, vb, o2[t]);
    }
  }

  // ---- combine, RMS-norm over 128 dims, scale, store bf16
  const float lamh = lam[h];
  float ss[8];
#pragma unroll
  for (int r = 0; r < 8; ++r) ss[r] = 0.f;
#pragma unroll
  for (int t = 0; t < 8; ++t)
#pragma unroll
    for (int r = 0; r < 8; ++r) {
      float v = o1[t][r] / l1[r] - lamh * (o2[t][r] / l2[r]);
      o1[t][r] = v;
      ss[r] += v * v;
    }
#pragma unroll
  for (int r = 0; r < 8; ++r) {
    float tot = redsum16(ss[r]);
    ss[r] = rsqrtf(tot * (1.0f / 128.0f) + 1e-5f);  // 1/rms
  }
#pragma unroll
  for (int t = 0; t < 8; ++t) {
    float rsc = rms_scale[t * 16 + lo] * 0.2f;  // * (1 - lambda_init)
#pragma unroll
    for (int r = 0; r < 8; ++r) {
      size_t row = (size_t)(b * NN + q0 + r + hi * 8);
      Ocat[row * D2 + h * 2 * DHEAD + t * 16 + lo] = (__bf16)(o1[t][r] * ss[r] * rsc);
    }
  }
}

// ---------------------------------------------------------------- launch
extern "C" void kernel_launch(void* const* d_in, const int* in_sizes, int n_in,
                              void* d_out, int out_size, void* d_ws, size_t ws_size,
                              hipStream_t stream) {
  const float* X   = (const float*)d_in[0];
  const float* Wq  = (const float*)d_in[1];
  const float* Wk  = (const float*)d_in[2];
  const float* Wv  = (const float*)d_in[3];
  const float* Wo  = (const float*)d_in[4];
  const float* lq1 = (const float*)d_in[5];
  const float* lk1 = (const float*)d_in[6];
  const float* lq2 = (const float*)d_in[7];
  const float* lk2 = (const float*)d_in[8];
  const float* rms = (const float*)d_in[9];

  char* ws = (char*)d_ws;
  size_t off = 0;
  __bf16* Xb  = (__bf16*)(ws + off); off += (size_t)BB * NN * DM * 2;        // 8 MB
  __bf16* Wqb = (__bf16*)(ws + off); off += (size_t)D2 * DM * 2;             // 4 MB
  __bf16* Wkb = (__bf16*)(ws + off); off += (size_t)D2 * DM * 2;
  __bf16* Wvb = (__bf16*)(ws + off); off += (size_t)D2 * DM * 2;
  __bf16* Wob = (__bf16*)(ws + off); off += (size_t)DM * D2 * 2;
  __bf16* Qp  = (__bf16*)(ws + off); off += (size_t)BB * NN * D2 * 2;        // 16 MB
  __bf16* Kp  = (__bf16*)(ws + off); off += (size_t)BB * NN * D2 * 2;
  __bf16* Vp  = (__bf16*)(ws + off); off += (size_t)BB * NN * D2 * 2;
  __bf16* Vt  = (__bf16*)(ws + off); off += (size_t)BB * NN * D2 * 2;
  __bf16* Oc  = (__bf16*)(ws + off); off += (size_t)BB * NN * D2 * 2;
  float*  lam = (float*)(ws + off);  off += 64;

  // fp32 -> bf16 converts
  int nX = BB * NN * DM;           // 4,194,304
  int nW = D2 * DM;                // 2,097,152
  f32_to_bf16_kernel<<<(nX + 255) / 256, 256, 0, stream>>>(X,  Xb,  nX);
  f32_to_bf16_kernel<<<(nW + 255) / 256, 256, 0, stream>>>(Wq, Wqb, nW);
  f32_to_bf16_kernel<<<(nW + 255) / 256, 256, 0, stream>>>(Wk, Wkb, nW);
  f32_to_bf16_kernel<<<(nW + 255) / 256, 256, 0, stream>>>(Wv, Wvb, nW);
  f32_to_bf16_kernel<<<(nW + 255) / 256, 256, 0, stream>>>(Wo, Wob, nW);
  lam_kernel<<<1, 32, 0, stream>>>(lq1, lk1, lq2, lk2, lam);

  // projections: [4096 x 1024] @ [2048 x 1024]^T -> bf16 [4096 x 2048]
  dim3 gp(D2 / 256, (BB * NN) / 128);  // (8, 32)
  gemm_nt_kernel<true><<<gp, 256, 0, stream>>>(Xb, Wqb, nullptr, Qp, BB * NN, D2, DM);
  gemm_nt_kernel<true><<<gp, 256, 0, stream>>>(Xb, Wkb, nullptr, Kp, BB * NN, D2, DM);
  gemm_nt_kernel<true><<<gp, 256, 0, stream>>>(Xb, Wvb, nullptr, Vp, BB * NN, D2, DM);

  // V transpose to [bh, 128, key]
  int nV = BB * NHEADS * 2 * DHEAD * NN;  // 8,388,608
  transpose_v_kernel<<<(nV + 255) / 256, 256, 0, stream>>>(Vp, Vt);

  // differential attention
  diff_attn_kernel<<<dim3(NN / 16, BB * NHEADS), 32, 0, stream>>>(Qp, Kp, Vt, lam, rms, Oc);

  // output projection: [4096 x 2048] @ [1024 x 2048]^T -> fp32 d_out
  dim3 go(DM / 256, (BB * NN) / 128);  // (4, 32)
  gemm_nt_kernel<false><<<go, 256, 0, stream>>>(Oc, Wob, (float*)d_out, nullptr,
                                                BB * NN, DM, D2);
}
